// MambaVisionMixer_v9_24206435680567
// MI455X (gfx1250) — matrix operand order repaired and checked by reading, not compiled, not run on hardware
//
#include <hip/hip_runtime.h>
#include <hip/hip_bf16.h>

typedef __attribute__((ext_vector_type(16))) _Float16 v16h;
typedef __attribute__((ext_vector_type(8)))  float    v8f;
typedef __attribute__((ext_vector_type(4)))  unsigned int v4u;
typedef __attribute__((ext_vector_type(4)))  int v4i;
typedef __attribute__((ext_vector_type(8)))  int v8i;

#define B_    4
#define M_    4
#define L_    1024
#define DM_   768
#define D2_   384
#define LM_   4096      // L_*M_ interleaved sequence
#define LR_   16
#define DTR_  48
#define DS_   8
#define XDBL_ 64        // DTR_+2*DS_

// ---------------- workspace layout (bytes) ----------------
constexpr size_t OFF_W1H   = 0;                                  // M*LR*DM f16
constexpr size_t OFF_W2H   = OFF_W1H  + (size_t)M_*LR_*DM_*2;    // M*DM*LR f16
constexpr size_t OFF_XPWH  = OFF_W2H  + (size_t)M_*DM_*LR_*2;    // 64*384 f16
constexpr size_t OFF_DTWH  = OFF_XPWH + (size_t)XDBL_*D2_*2;     // 384*48 f16
constexpr size_t OFF_OUTWH = OFF_DTWH + (size_t)D2_*DTR_*2;      // M*DM*D2 f16
constexpr size_t OFF_XZ    = ((OFF_OUTWH + (size_t)M_*DM_*D2_*2) + 255) & ~size_t(255); // B*LM*DM f32
constexpr size_t OFF_XC32  = OFF_XZ    + (size_t)B_*LM_*DM_*4;   // B*LM*D2 f32
constexpr size_t OFF_XCH   = OFF_XC32  + (size_t)B_*LM_*D2_*4;   // B*LM*D2 f16
constexpr size_t OFF_ZC32  = OFF_XCH   + (size_t)B_*LM_*D2_*2;   // B*LM*D2 f32
constexpr size_t OFF_DELTA = OFF_ZC32  + (size_t)B_*LM_*D2_*4;   // B*LM*D2 f32
constexpr size_t OFF_BCCC  = OFF_DELTA + (size_t)B_*LM_*D2_*4;   // B*LM*16 f32
constexpr size_t OFF_Y     = OFF_BCCC  + (size_t)B_*LM_*16*4;    // B*LM*D2 f32
constexpr size_t OFF_YGH   = OFF_Y     + (size_t)B_*LM_*D2_*4;   // B*LM*D2 f16

static __device__ __forceinline__ float siluf(float v) { return v / (1.0f + __expf(-v)); }
static __device__ __forceinline__ float softplusf(float v) {
    return (v > 20.0f) ? v : log1pf(__expf(v));
}

// ---------------- weight f32 -> f16 ----------------
__global__ void k_cvt_f16(const float* __restrict__ s, _Float16* __restrict__ d, int n) {
    int i = blockIdx.x * blockDim.x + threadIdx.x;
    if (i < n) d[i] = (_Float16)s[i];
}

// ---------------- kernel 1: fused in-projection ----------------
// grid (B*M, L/16), block 32 (1 wave). Per block: 16 tokens of one (b, modality).
__global__ void k_inproj(const float* __restrict__ hs,
                         const float* __restrict__ ln_g, const float* __restrict__ ln_b,
                         const _Float16* __restrict__ w1h, const _Float16* __restrict__ w2h,
                         const int* __restrict__ midx,
                         float* __restrict__ xz) {
    __shared__ _Float16 hA[16 * DM_];   // 16 tokens x 768, f16 (24 KB)
    __shared__ _Float16 shT[16 * 32];   // t matrix, K padded 16->32

    const int bm   = blockIdx.x;
    const int b    = bm >> 2;
    const int mi   = bm & 3;
    const int mm   = midx[mi];
    const int l0   = blockIdx.y * 16;
    const int lane = threadIdx.x;

    // stage 16x768 f32 tile -> f16 LDS (coalesced)
    const size_t hbase = (((size_t)b * M_ + mi) * L_ + l0) * DM_;
    for (int i = lane; i < 16 * DM_; i += 32) hA[i] = (_Float16)hs[hbase + i];
    for (int i = lane; i < 16 * 32; i += 32) shT[i] = (_Float16)0.0f;
    __syncthreads();

    const int r  = lane & 15;
    const int hi = lane >> 4;

    // GEMM1: t(16x16) = h(16x768) x w1^T(768x16)
    v8f acc = {};
    for (int k0 = 0; k0 < DM_; k0 += 32) {
        v16h a, bb;
        const _Float16* pa0 = &hA[r * DM_ + k0 + hi * 16];
        const _Float16* pb  = w1h + ((size_t)(mm * LR_ + r)) * DM_ + k0 + hi * 16;
#pragma unroll
        for (int j = 0; j < 8; ++j) { a[j] = pa0[j]; a[8 + j] = pa0[8 + j]; }
#pragma unroll
        for (int j = 0; j < 16; ++j) bb[j] = pb[j];
        acc = __builtin_amdgcn_wmma_f32_16x16x32_f16(false, a, false, bb, (short)0, acc, false, false);
    }

    // LayerNorm over rank dim (16 lanes of each half-group hold one row) + GELU
    const float g  = ln_g[mm * LR_ + r];
    const float bt = ln_b[mm * LR_ + r];
#pragma unroll
    for (int v = 0; v < 8; ++v) {
        float x  = acc[v];
        float s1 = x, s2 = x * x;
#pragma unroll
        for (int msk = 1; msk < 16; msk <<= 1) {
            s1 += __shfl_xor(s1, msk, 32);
            s2 += __shfl_xor(s2, msk, 32);
        }
        float mu  = s1 * (1.0f / 16.0f);
        float var = s2 * (1.0f / 16.0f) - mu * mu;
        float t   = (x - mu) * rsqrtf(var + 1e-5f) * g + bt;
        t = 0.5f * t * (1.0f + erff(t * 0.70710678118654752f));   // exact GELU
        shT[(v + 8 * hi) * 32 + r] = (_Float16)t;                 // row=token, col=rank
    }
    __syncthreads();

    // GEMM2: xz(16x768) = t(16x16, K padded to 32) x w2^T(16x768)
    v16h a2;
    {
        const _Float16* pa0 = &shT[r * 32 + hi * 16];
#pragma unroll
        for (int j = 0; j < 8; ++j) { a2[j] = pa0[j]; a2[8 + j] = pa0[8 + j]; }
    }
    const size_t xzb = (size_t)b * LM_ * DM_;
    for (int nt = 0; nt < DM_ / 16; ++nt) {
        v16h b2;
        if (hi == 0) {
            const _Float16* pb = w2h + ((size_t)mm * DM_ + nt * 16 + r) * LR_;
#pragma unroll
            for (int j = 0; j < 16; ++j) b2[j] = pb[j];
        } else {
#pragma unroll
            for (int j = 0; j < 16; ++j) b2[j] = (_Float16)0.0f;   // K=16..31 pad
        }
        v8f c2 = {};
        c2 = __builtin_amdgcn_wmma_f32_16x16x32_f16(false, a2, false, b2, (short)0, c2, false, false);
        const int dcol = nt * 16 + r;
#pragma unroll
        for (int v = 0; v < 8; ++v) {
            const int l = l0 + v + 8 * hi;
            const int s = l * M_ + mi;                 // modality interleave
            xz[xzb + (size_t)s * DM_ + dcol] = c2[v];
        }
    }
}

// ---------------- kernel 2: depthwise conv k=3 (SAME) + SiLU ----------------
__global__ void k_conv(const float* __restrict__ xz,
                       const float* __restrict__ wx, const float* __restrict__ wz,
                       float* __restrict__ xc32, _Float16* __restrict__ xch,
                       float* __restrict__ zc32) {
    const int n = B_ * LM_ * D2_;
    for (int idx = blockIdx.x * blockDim.x + threadIdx.x; idx < n; idx += gridDim.x * blockDim.x) {
        const int d = idx % D2_;
        const int s = (idx / D2_) % LM_;
        const int b = idx / (D2_ * LM_);
        const size_t base = ((size_t)b * LM_) * DM_;
        const size_t p = base + (size_t)s * DM_;
        float xm = (s > 0)       ? xz[p - DM_ + d] : 0.0f;
        float x0 = xz[p + d];
        float xp = (s < LM_ - 1) ? xz[p + DM_ + d] : 0.0f;
        float zm = (s > 0)       ? xz[p - DM_ + D2_ + d] : 0.0f;
        float z0 = xz[p + D2_ + d];
        float zp = (s < LM_ - 1) ? xz[p + DM_ + D2_ + d] : 0.0f;
        float xv = siluf(wx[d * 3] * xm + wx[d * 3 + 1] * x0 + wx[d * 3 + 2] * xp);
        float zv = siluf(wz[d * 3] * zm + wz[d * 3 + 1] * z0 + wz[d * 3 + 2] * zp);
        xc32[idx] = xv;
        xch[idx]  = (_Float16)xv;
        zc32[idx] = zv;
    }
}

// ---------------- kernel 3: x_proj + dt_proj (WMMA) + softplus ----------------
// grid (B, LM/16), block 32. 96 WMMAs per block.
__global__ void k_xdbl(const _Float16* __restrict__ xch,
                       const _Float16* __restrict__ xpwh, const _Float16* __restrict__ dtwh,
                       const float* __restrict__ dt_b,
                       float* __restrict__ bccc, float* __restrict__ delta) {
    __shared__ _Float16 shD[16 * 64];   // dt-input 16x48 padded to 16x64

    const int b    = blockIdx.x;
    const int s0   = blockIdx.y * 16;
    const int lane = threadIdx.x;
    const int r    = lane & 15;
    const int hi   = lane >> 4;

    for (int i = lane; i < 16 * 64; i += 32) shD[i] = (_Float16)0.0f;
    __syncthreads();

    // x_dbl(16x64) = xc(16x384) x x_proj_w^T(384x64)
    v8f accs[4] = {v8f{}, v8f{}, v8f{}, v8f{}};
    const size_t abase = ((size_t)b * LM_ + s0) * D2_;
    for (int k0 = 0; k0 < D2_; k0 += 32) {
        v16h a;
        const _Float16* pa0 = xch + abase + (size_t)r * D2_ + k0 + hi * 16;
#pragma unroll
        for (int j = 0; j < 8; ++j) { a[j] = pa0[j]; a[8 + j] = pa0[8 + j]; }
#pragma unroll
        for (int nt = 0; nt < 4; ++nt) {
            v16h bb;
            const _Float16* pb = xpwh + ((size_t)(nt * 16 + r)) * D2_ + k0 + hi * 16;
#pragma unroll
            for (int j = 0; j < 16; ++j) bb[j] = pb[j];
            accs[nt] = __builtin_amdgcn_wmma_f32_16x16x32_f16(false, a, false, bb, (short)0, accs[nt], false, false);
        }
    }

    // cols 48..63 -> Bc/Cc; cols 0..47 -> LDS (f16) for dt GEMM
    const size_t bcb = ((size_t)b * LM_ + s0) * 16;
#pragma unroll
    for (int v = 0; v < 8; ++v) {
        const int m = v + 8 * hi;
        bccc[bcb + (size_t)m * 16 + r] = accs[3][v];
        shD[m * 64 + 0 * 16 + r]  = (_Float16)accs[0][v];
        shD[m * 64 + 1 * 16 + r]  = (_Float16)accs[1][v];
        shD[m * 64 + 2 * 16 + r]  = (_Float16)accs[2][v];
    }
    __syncthreads();

    // dt(16x384) = x_dbl[:, :48](pad 64) x dt_proj_w^T(48x384); delta = softplus(dt + 2*bias)
    const size_t dbase = ((size_t)b * LM_ + s0) * D2_;
    for (int nt = 0; nt < D2_ / 16; ++nt) {
        v8f c2 = {};
        const int dcol = nt * 16 + r;
#pragma unroll
        for (int k0 = 0; k0 < 64; k0 += 32) {
            v16h a2, b2;
            const _Float16* pa0 = &shD[r * 64 + k0 + hi * 16];
#pragma unroll
            for (int j = 0; j < 8; ++j) { a2[j] = pa0[j]; a2[8 + j] = pa0[8 + j]; }
#pragma unroll
            for (int j = 0; j < 16; ++j) {
                const int k = k0 + hi * 16 + j;
                b2[j] = (k < DTR_) ? dtwh[(size_t)dcol * DTR_ + k] : (_Float16)0.0f;
            }
            c2 = __builtin_amdgcn_wmma_f32_16x16x32_f16(false, a2, false, b2, (short)0, c2, false, false);
        }
        const float b2x = 2.0f * dt_b[dcol];
#pragma unroll
        for (int v = 0; v < 8; ++v) {
            const int m = v + 8 * hi;
            delta[dbase + (size_t)m * D2_ + dcol] = softplusf(c2[v] + b2x);
        }
    }
}

// ---------------- kernel 4: selective scan ----------------
// grid (B, 3), block 128: one thread per channel d, 8 states in registers.
__global__ void k_scan(const float* __restrict__ delta, const float* __restrict__ xc32,
                       const float* __restrict__ bccc,
                       const float* __restrict__ A_log, const float* __restrict__ Dp,
                       float* __restrict__ y) {
    __shared__ float shBC[128][16];
    const int b   = blockIdx.x;
    const int d   = blockIdx.y * 128 + threadIdx.x;
    const int tid = threadIdx.x;

    float Ac[DS_];
#pragma unroll
    for (int j = 0; j < DS_; ++j) Ac[j] = -__expf(A_log[d * DS_ + j]);
    const float Dd = Dp[d];
    float h[DS_];
#pragma unroll
    for (int j = 0; j < DS_; ++j) h[j] = 0.0f;

    const size_t dxb = (size_t)b * LM_ * D2_;
    const size_t bcb = (size_t)b * LM_ * 16;

    for (int t0 = 0; t0 < LM_; t0 += 128) {
        __syncthreads();
        for (int i = tid; i < 128 * 16; i += 128)
            shBC[i >> 4][i & 15] = bccc[bcb + (size_t)(t0 + (i >> 4)) * 16 + (i & 15)];
        __syncthreads();
        for (int tt = 0; tt < 128; ++tt) {
            const int s   = t0 + tt;
            const float dv = delta[dxb + (size_t)s * D2_ + d];
            const float xv = xc32[dxb + (size_t)s * D2_ + d];
            float acc = Dd * xv;
            const float dx = dv * xv;
#pragma unroll
            for (int j = 0; j < DS_; ++j) {
                const float dA = __expf(dv * Ac[j]);
                h[j] = dA * h[j] + dx * shBC[tt][j];         // Bc = cols 0..7
                acc += h[j] * shBC[tt][8 + j];               // Cc = cols 8..15
            }
            y[dxb + (size_t)s * D2_ + d] = acc;
        }
    }
}

// ---------------- kernel 5: RMSNorm + SiLU gate -> f16 ----------------
__global__ void k_normgate(const float* __restrict__ y, const float* __restrict__ zc32,
                           const float* __restrict__ nw, _Float16* __restrict__ ygh) {
    __shared__ float red[128];
    const int tok = blockIdx.x;
    const int b   = tok >> 12;
    const int s   = tok & (LM_ - 1);
    const int tid = threadIdx.x;
    const size_t base = ((size_t)b * LM_ + s) * D2_;

    float ss = 0.0f, vv[3];
#pragma unroll
    for (int i = 0; i < 3; ++i) { vv[i] = y[base + tid + i * 128]; ss += vv[i] * vv[i]; }
    red[tid] = ss;
    __syncthreads();
    for (int off = 64; off > 0; off >>= 1) {
        if (tid < off) red[tid] += red[tid + off];
        __syncthreads();
    }
    const float rms = rsqrtf(red[0] * (1.0f / D2_) + 1e-5f);
#pragma unroll
    for (int i = 0; i < 3; ++i) {
        const int d = tid + i * 128;
        const float g = siluf(zc32[base + d]);
        ygh[base + d] = (_Float16)(vv[i] * rms * nw[d] * g);
    }
}

// ---------------- kernel 6: out-projection (dominant GEMM, WMMA + TDM) ----------------
// grid (B*M, L/16, DM/64), block 32. A tile (16 tok x 384, row stride 4*384) staged to
// LDS via the Tensor Data Mover; 4 N-tiles share one A fragment per K-step.
__global__ void k_outproj(const _Float16* __restrict__ ygh, const _Float16* __restrict__ wh,
                          const float* __restrict__ scale_p, const int* __restrict__ midx,
                          float* __restrict__ out) {
    __shared__ _Float16 shA[16 * D2_];   // 12 KB staged activation tile

    const int bm   = blockIdx.x;
    const int b    = bm >> 2;
    const int mi   = bm & 3;
    const int mm   = midx[mi];
    const int l0   = blockIdx.y * 16;
    const int o0   = blockIdx.z * 64;
    const int lane = threadIdx.x;
    const int r    = lane & 15;
    const int hi   = lane >> 4;

    const size_t gelem = ((size_t)b * LM_ + (size_t)l0 * M_ + mi) * D2_;  // first row, f16 elems
#if defined(__has_builtin) && __has_builtin(__builtin_amdgcn_tensor_load_to_lds) && __has_builtin(__builtin_amdgcn_s_wait_tensorcnt)
    {
        // Tensor DMA descriptor: 2D tile, 16 rows x 384 f16, row stride 1536 elements.
        const unsigned long long ga = (unsigned long long)(uintptr_t)(ygh + gelem);
        const unsigned lds_off = (unsigned)(uintptr_t)(void*)&shA[0];
        v4u g0; v8i g1;
        g0[0] = 1u;                                               // count=1 (valid user D#)
        g0[1] = lds_off;                                          // lds_addr (bytes)
        g0[2] = (unsigned)(ga & 0xFFFFFFFFu);                     // global_addr[31:0]
        g0[3] = (unsigned)((ga >> 32) & 0x1FFFFFFu) | (2u << 30); // global_addr[56:32] | type=2
        g1[0] = (int)(1u << 16);         // workgroup_mask=0 | data_size=1 (2 bytes)
        g1[1] = (int)(384u << 16);       // tensor_dim0 = 384 (low 16)
        g1[2] = (int)(16u << 16);        // tensor_dim0 hi=0 | tensor_dim1 = 16 (low 16)
        g1[3] = (int)(384u << 16);       // tensor_dim1 hi=0 | tile_dim0 = 384
        g1[4] = (int)16u;                // tile_dim1 = 16 | tile_dim2 = 0
        g1[5] = (int)(D2_ * M_);         // tensor_dim0_stride = 1536 (low 32)
        g1[6] = 0;                       // stride hi | tensor_dim1_stride lo
        g1[7] = 0;
        v4i z4 = {};
#if __clang_major__ >= 23
        v8i z8 = {};
        __builtin_amdgcn_tensor_load_to_lds(g0, g1, z4, z4, z8, 0);
#else
        __builtin_amdgcn_tensor_load_to_lds(g0, g1, z4, z4, 0);
#endif
        __builtin_amdgcn_s_wait_tensorcnt(0);
    }
#else
    for (int i = lane; i < 16 * D2_; i += 32) {
        const int row = i / D2_, col = i % D2_;
        shA[i] = ygh[gelem + (size_t)row * (D2_ * M_) + col];
    }
#endif
    __syncthreads();

    v8f acc[4] = {v8f{}, v8f{}, v8f{}, v8f{}};
    for (int k0 = 0; k0 < D2_; k0 += 32) {
        v16h a;
        const _Float16* pa0 = &shA[r * D2_ + k0 + hi * 16];
#pragma unroll
        for (int j = 0; j < 8; ++j) { a[j] = pa0[j]; a[8 + j] = pa0[8 + j]; }
#pragma unroll
        for (int nt = 0; nt < 4; ++nt) {
            v16h bb;
            const _Float16* pb = wh + ((size_t)mm * DM_ + o0 + nt * 16 + r) * D2_ + k0 + hi * 16;
            if (k0 + 32 < D2_) __builtin_prefetch(pb + 32, 0, 0);   // global_prefetch_b8
#pragma unroll
            for (int j = 0; j < 16; ++j) bb[j] = pb[j];
            acc[nt] = __builtin_amdgcn_wmma_f32_16x16x32_f16(false, a, false, bb, (short)0, acc[nt], false, false);
        }
    }
    const float sc = scale_p[0];
    const size_t ob = (((size_t)b * M_ + mi) * L_) * DM_;
#pragma unroll
    for (int nt = 0; nt < 4; ++nt) {
#pragma unroll
        for (int v = 0; v < 8; ++v) {
            const int l = l0 + v + 8 * hi;
            out[ob + (size_t)l * DM_ + o0 + nt * 16 + r] = acc[nt][v] * sc;
        }
    }
}

// ---------------- host launch ----------------
extern "C" void kernel_launch(void* const* d_in, const int* in_sizes, int n_in,
                              void* d_out, int out_size, void* d_ws, size_t ws_size,
                              hipStream_t stream) {
    const float* hs    = (const float*)d_in[0];
    const float* in_w1 = (const float*)d_in[1];
    const float* ln_g  = (const float*)d_in[2];
    const float* ln_b  = (const float*)d_in[3];
    const float* in_w2 = (const float*)d_in[4];
    const float* xpw   = (const float*)d_in[5];
    const float* dtw   = (const float*)d_in[6];
    const float* dt_b  = (const float*)d_in[7];
    const float* A_log = (const float*)d_in[8];
    const float* Dp    = (const float*)d_in[9];
    const float* cwx   = (const float*)d_in[10];
    const float* cwz   = (const float*)d_in[11];
    const float* nw    = (const float*)d_in[12];
    const float* outw  = (const float*)d_in[13];
    const float* scale = (const float*)d_in[14];
    const int*   midx  = (const int*)d_in[15];
    float* out = (float*)d_out;

    char* ws = (char*)d_ws;
    _Float16* w1h  = (_Float16*)(ws + OFF_W1H);
    _Float16* w2h  = (_Float16*)(ws + OFF_W2H);
    _Float16* xpwh = (_Float16*)(ws + OFF_XPWH);
    _Float16* dtwh = (_Float16*)(ws + OFF_DTWH);
    _Float16* owh  = (_Float16*)(ws + OFF_OUTWH);
    float*    xz   = (float*)(ws + OFF_XZ);
    float*    xc32 = (float*)(ws + OFF_XC32);
    _Float16* xch  = (_Float16*)(ws + OFF_XCH);
    float*    zc32 = (float*)(ws + OFF_ZC32);
    float*    delta= (float*)(ws + OFF_DELTA);
    float*    bccc = (float*)(ws + OFF_BCCC);
    float*    y    = (float*)(ws + OFF_Y);
    _Float16* ygh  = (_Float16*)(ws + OFF_YGH);

    auto cvt = [&](const float* s, _Float16* d, int n) {
        k_cvt_f16<<<(n + 255) / 256, 256, 0, stream>>>(s, d, n);
    };
    cvt(in_w1, w1h,  M_ * LR_ * DM_);
    cvt(in_w2, w2h,  M_ * DM_ * LR_);
    cvt(xpw,   xpwh, XDBL_ * D2_);
    cvt(dtw,   dtwh, D2_ * DTR_);
    cvt(outw,  owh,  M_ * DM_ * D2_);

    k_inproj<<<dim3(B_ * M_, L_ / 16), 32, 0, stream>>>(hs, ln_g, ln_b, w1h, w2h, midx, xz);

    k_conv<<<(B_ * LM_ * D2_ + 255) / 256, 256, 0, stream>>>(xz, cwx, cwz, xc32, xch, zc32);

    k_xdbl<<<dim3(B_, LM_ / 16), 32, 0, stream>>>(xch, xpwh, dtwh, dt_b, bccc, delta);

    k_scan<<<dim3(B_, D2_ / 128), 128, 0, stream>>>(delta, xc32, bccc, A_log, Dp, y);

    k_normgate<<<B_ * LM_, 128, 0, stream>>>(y, zc32, nw, ygh);

    k_outproj<<<dim3(B_ * M_, L_ / 16, DM_ / 64), 32, 0, stream>>>(ygh, owh, scale, midx, out);
}
